// MultiScaleAttention_25348896981132
// MI455X (gfx1250) — matrix operand order repaired and checked by reading, not compiled, not run on hardware
//
#include <hip/hip_runtime.h>
#include <hip/hip_bf16.h>

typedef __attribute__((ext_vector_type(4)))  __bf16 v4bf;
typedef __attribute__((ext_vector_type(8)))  __bf16 v8bf;
typedef __attribute__((ext_vector_type(16))) __bf16 v16bf;
typedef __attribute__((ext_vector_type(8)))  float  v8f;

static __device__ inline v8f vzero8f() {
    v8f z;
#pragma unroll
    for (int i = 0; i < 8; ++i) z[i] = 0.0f;
    return z;
}

// Build a 16-element bf16 A fragment from two 8-element (16B) LDS chunks.
static __device__ inline v16bf make_frag2(const __bf16* p0, const __bf16* p1) {
    union { v16bf v; v8bf h[2]; } u;
    u.h[0] = *(const v8bf*)p0;
    u.h[1] = *(const v8bf*)p1;
    return u.v;
}

// CDNA5 async global->LDS copy (16 bytes per lane), tracked by ASYNCcnt.
static __device__ inline void async_copy_b128(void* lds_dst, const void* gsrc) {
    unsigned l = (unsigned)(uintptr_t)lds_dst;               // low 32 bits = LDS offset
    unsigned long long g = (unsigned long long)(uintptr_t)gsrc;
    asm volatile("global_load_async_to_lds_b128 %0, %1, off"
                 :: "v"(l), "v"(g) : "memory");
}
static __device__ inline void wait_async0() {
    asm volatile("s_wait_asynccnt 0x0" ::: "memory");
}

// ---------------------------------------------------------------------------
// Cast fp32 -> bf16, 4 elements/thread
// ---------------------------------------------------------------------------
__global__ __launch_bounds__(256) void cast_f32_to_bf16_x4(
    const float4* __restrict__ in, __bf16* __restrict__ out, long n4) {
    long i = (long)blockIdx.x * 256 + threadIdx.x;
    if (i >= n4) return;
    float4 v = in[i];
    v4bf o;
    o[0] = (__bf16)v.x; o[1] = (__bf16)v.y; o[2] = (__bf16)v.z; o[3] = (__bf16)v.w;
    *(v4bf*)(out + i * 4) = o;
}

// ---------------------------------------------------------------------------
// Transpose + cast:  W[R][C] fp32 -> Wt[C][R] bf16  (small weight matrices)
// ---------------------------------------------------------------------------
__global__ __launch_bounds__(256) void transpose_cast_kernel(
    const float* __restrict__ W, __bf16* __restrict__ Wt, int R, int C) {
    long i = (long)blockIdx.x * 256 + threadIdx.x;
    if (i >= (long)R * C) return;
    int r = (int)(i / C);
    int c = (int)(i % C);
    Wt[(size_t)c * R + r] = (__bf16)W[i];
}

// ---------------------------------------------------------------------------
// bf16 GEMM:  C[M][N] = A[M][K] @ Bt[N][K]^T + bias[N]
// Block tile 128M x 256N, 8 waves as 2(M) x 4(N); each wave owns a 64x64 tile
// (4x4 WMMA 16x16 tiles -> 16 v_wmma per K-step of 32).
// K staged 32 at a time, double-buffered via async global->LDS (ASYNCcnt).
// M mult of 128, N mult of 256, K mult of 32.
// ---------------------------------------------------------------------------
template <bool BF16_OUT>
__global__ __launch_bounds__(256) void gemm_bf16_kernel(
    const __bf16* __restrict__ A, const __bf16* __restrict__ Bt,
    const float* __restrict__ bias, void* __restrict__ Cout,
    int M, int N, int K) {
    __shared__ __align__(32) __bf16 as[2][128 * 32];   // 2 x 8KB
    __shared__ __align__(32) __bf16 bs[2][256 * 32];   // 2 x 16KB

    const int tid   = threadIdx.x;
    const int lane  = tid & 31;
    const int wave  = tid >> 5;
    const int waveM = wave & 1;    // 0..1 -> M offset waveM*64
    const int waveN = wave >> 1;   // 0..3 -> N offset waveN*64
    const int lm    = lane & 15;
    const int hh    = lane >> 4;

    const int blockN = blockIdx.x * 256;
    const int blockM = blockIdx.y * 128;

    v8f acc[4][4];
#pragma unroll
    for (int mt = 0; mt < 4; ++mt)
#pragma unroll
        for (int nt = 0; nt < 4; ++nt) acc[mt][nt] = vzero8f();

    // Async stage of one 32-wide K slice into buffer `buf`.
    auto stage = [&](int kt, int buf) {
#pragma unroll
        for (int i = 0; i < 2; ++i) {               // A: 128 rows x 4 chunks
            int idx = tid + 256 * i;
            int r = idx >> 2;
            int c = (idx & 3) << 3;
            async_copy_b128(&as[buf][r * 32 + c],
                            &A[(size_t)(blockM + r) * K + kt + c]);
        }
#pragma unroll
        for (int i = 0; i < 4; ++i) {               // B: 256 rows x 4 chunks
            int idx = tid + 256 * i;
            int r = idx >> 2;
            int c = (idx & 3) << 3;
            async_copy_b128(&bs[buf][r * 32 + c],
                            &Bt[(size_t)(blockN + r) * K + kt + c]);
        }
    };

    stage(0, 0);
    wait_async0();
    __syncthreads();

    const int nStages = K >> 5;
    for (int s = 0; s < nStages; ++s) {
        const int cur = s & 1;
        if (s + 1 < nStages) stage((s + 1) << 5, cur ^ 1);   // prefetch next slice

        v16bf afrag[4];
#pragma unroll
        for (int mt = 0; mt < 4; ++mt) {
            // A layout: lanes 0-15 hold K 0-15; lanes 16-31 hold K 16-31
            const __bf16* ap = &as[cur][(waveM * 64 + mt * 16 + lm) * 32 + hh * 16];
            afrag[mt] = make_frag2(ap, ap + 8);
        }
        v16bf bfrag[4];
#pragma unroll
        for (int nt = 0; nt < 4; ++nt) {
            // B layout: lane holds 16 consecutive K of its column (K-major LDS)
            bfrag[nt] = *(const v16bf*)&bs[cur][(waveN * 64 + nt * 16 + lm) * 32 + hh * 16];
        }
#pragma unroll
        for (int mt = 0; mt < 4; ++mt)
#pragma unroll
            for (int nt = 0; nt < 4; ++nt)
                acc[mt][nt] = __builtin_amdgcn_wmma_f32_16x16x32_bf16(
                    false, afrag[mt], false, bfrag[nt],
                    (short)0, acc[mt][nt], false, false);

        wait_async0();
        __syncthreads();
    }

    // Epilogue: bias add + store. D layout: lanes 0-15 VGPR r -> row r; lanes 16-31 -> row r+8.
#pragma unroll
    for (int mt = 0; mt < 4; ++mt)
#pragma unroll
        for (int nt = 0; nt < 4; ++nt) {
            int col = blockN + waveN * 64 + nt * 16 + lm;
            float bv = bias[col];
#pragma unroll
            for (int r = 0; r < 8; ++r) {
                int row = blockM + waveM * 64 + mt * 16 + r + 8 * hh;
                float val = acc[mt][nt][r] + bv;
                if (BF16_OUT)
                    ((__bf16*)Cout)[(size_t)row * N + col] = (__bf16)val;
                else
                    ((float*)Cout)[(size_t)row * N + col] = val;
            }
        }
}

// ---------------------------------------------------------------------------
// Windowed attention, one wave32 per (b, h, w).
// qkv: [B][3136][2304] bf16; row n = t*49 + w; cols = {q,k,v}*768 + h*64 + d.
// Pooled q[qi] = max_g q[g*16+qi]. logits = qp@K^T * 1/8; softmax; o = P@V.
// obuf layout: [B][16 qi][49 w][12 h][64 d] bf16  (== [B, 784, 768] for proj GEMM)
// ---------------------------------------------------------------------------
__global__ __launch_bounds__(32) void attn_kernel(
    const __bf16* __restrict__ qkv, __bf16* __restrict__ obuf) {
    __shared__ __align__(32) __bf16 ksm[64 * 64];   // K rows [t][d]
    __shared__ __align__(32) __bf16 vtm[64 * 64];   // V transposed [d][t]
    __shared__ __align__(32) __bf16 qsm[16 * 64];   // pooled q [qi][d]
    __shared__ __align__(32) float  lsm[16 * 64];   // logits / exp
    __shared__ __align__(32) __bf16 psm[16 * 64];   // softmax probs bf16

    const int lane = threadIdx.x;
    const int lm = lane & 15;
    const int hh = lane >> 4;

    int idx = blockIdx.x;
    int w = idx % 49;
    int t1 = idx / 49;
    int h = t1 % 12;
    int b = t1 / 12;

    const size_t rowPitch = 49ull * 2304ull;  // token t stride in elements
    const __bf16* base = qkv + ((size_t)b * 3136 + (size_t)w) * 2304 + h * 64;

    // Stage K rows via async global->LDS (lane handles t = lane, lane+32)
#pragma unroll
    for (int rep = 0; rep < 2; ++rep) {
        int t = lane + rep * 32;
        const __bf16* src = base + (size_t)t * rowPitch + 768;
#pragma unroll
        for (int c = 0; c < 8; ++c)
            async_copy_b128(&ksm[t * 64 + c * 8], &src[c * 8]);
    }
    // Stage V transposed (needs VGPR round-trip for the transpose)
#pragma unroll
    for (int rep = 0; rep < 2; ++rep) {
        int t = lane + rep * 32;
        const __bf16* src = base + (size_t)t * rowPitch + 1536;
#pragma unroll
        for (int c = 0; c < 8; ++c) {
            v8bf tmp = *(const v8bf*)&src[c * 8];
#pragma unroll
            for (int j = 0; j < 8; ++j) vtm[(c * 8 + j) * 64 + t] = tmp[j];
        }
    }
    // Fused strided q max-pool: lane covers (qi=lm, d in [hh*32, hh*32+32))
    {
        int qi = lm;
        int dbase = hh * 32;
#pragma unroll
        for (int cc = 0; cc < 4; ++cc) {
            int d = dbase + cc * 8;
            float m[8];
#pragma unroll
            for (int j = 0; j < 8; ++j) m[j] = -3.4e38f;
#pragma unroll
            for (int g = 0; g < 4; ++g) {
                v8bf tq = *(const v8bf*)&base[(size_t)(g * 16 + qi) * rowPitch + d];
#pragma unroll
                for (int j = 0; j < 8; ++j) m[j] = fmaxf(m[j], (float)tq[j]);
            }
#pragma unroll
            for (int j = 0; j < 8; ++j) qsm[qi * 64 + d + j] = (__bf16)m[j];
        }
    }
    wait_async0();
    __syncthreads();

    // logits = qp @ K^T  (M=16, N=64 k-rows, K=64 dims)
    const float scale = 0.125f;  // 64^-0.5
#pragma unroll
    for (int nt = 0; nt < 4; ++nt) {
        v8f acc = vzero8f();
#pragma unroll
        for (int ks = 0; ks < 64; ks += 32) {
            const __bf16* ap = &qsm[lm * 64 + ks + hh * 16];
            v16bf a = make_frag2(ap, ap + 8);
            v16bf bb = *(const v16bf*)&ksm[(nt * 16 + lm) * 64 + ks + hh * 16];
            acc = __builtin_amdgcn_wmma_f32_16x16x32_bf16(
                false, a, false, bb, (short)0, acc, false, false);
        }
#pragma unroll
        for (int r = 0; r < 8; ++r)
            lsm[(r + 8 * hh) * 64 + nt * 16 + lm] = acc[r] * scale;
    }
    __syncthreads();

    // Softmax over 64 keys, one row per lane (lanes 0-15)
    if (lane < 16) {
        float mx = -3.4e38f;
        for (int j = 0; j < 64; ++j) mx = fmaxf(mx, lsm[lane * 64 + j]);
        float s = 0.0f;
        for (int j = 0; j < 64; ++j) {
            float e = __expf(lsm[lane * 64 + j] - mx);
            s += e;
            lsm[lane * 64 + j] = e;
        }
        float inv = 1.0f / s;
        for (int j = 0; j < 64; ++j)
            psm[lane * 64 + j] = (__bf16)(lsm[lane * 64 + j] * inv);
    }
    __syncthreads();

    // o = P @ V  (M=16, N=64 dims, K=64 keys), store in [B][qi][w][h][d]
#pragma unroll
    for (int nt = 0; nt < 4; ++nt) {
        v8f acc = vzero8f();
#pragma unroll
        for (int ks = 0; ks < 64; ks += 32) {
            const __bf16* ap = &psm[lm * 64 + ks + hh * 16];
            v16bf a = make_frag2(ap, ap + 8);
            v16bf bb = *(const v16bf*)&vtm[(nt * 16 + lm) * 64 + ks + hh * 16];
            acc = __builtin_amdgcn_wmma_f32_16x16x32_bf16(
                false, a, false, bb, (short)0, acc, false, false);
        }
        int d = nt * 16 + lm;
#pragma unroll
        for (int r = 0; r < 8; ++r) {
            int qi = r + 8 * hh;
            obuf[((((size_t)b * 16 + qi) * 49 + w) * 12 + h) * 64 + d] = (__bf16)acc[r];
        }
    }
}

// ---------------------------------------------------------------------------
// Host-side launch
// ---------------------------------------------------------------------------
extern "C" void kernel_launch(void* const* d_in, const int* in_sizes, int n_in,
                              void* d_out, int out_size, void* d_ws, size_t ws_size,
                              hipStream_t stream) {
    (void)in_sizes; (void)n_in; (void)out_size; (void)ws_size;

    const float* x     = (const float*)d_in[0];  // [32, 3136, 768]
    const float* Wqkv  = (const float*)d_in[1];  // [768, 2304]
    const float* bqkv  = (const float*)d_in[2];  // [2304]
    const float* Wproj = (const float*)d_in[3];  // [768, 768]
    const float* bproj = (const float*)d_in[4];  // [768]
    float*       out   = (float*)d_out;          // [32, 784, 768]

    char* ws = (char*)d_ws;
    // Workspace layout (bytes):
    __bf16* xb     = (__bf16*)(ws);               // 100352*768*2   = 154,140,672
    __bf16* wqkvT  = (__bf16*)(ws + 154140672);   // 2304*768*2     =   3,538,944
    __bf16* wprojT = (__bf16*)(ws + 157679616);   // 768*768*2      =   1,179,648
    __bf16* qkvb   = (__bf16*)(ws + 158859264);   // 100352*2304*2  = 462,422,016
    __bf16* obuf   = (__bf16*)(ws + 621281280);   // 25088*768*2    =  38,535,168

    // 1) x -> bf16  (77,070,336 elements = 19,267,584 float4)
    cast_f32_to_bf16_x4<<<75264, 256, 0, stream>>>((const float4*)x, xb, 19267584L);
    // 2) weights -> transposed bf16 (K-major for WMMA B operand)
    transpose_cast_kernel<<<6912, 256, 0, stream>>>(Wqkv, wqkvT, 768, 2304);
    transpose_cast_kernel<<<2304, 256, 0, stream>>>(Wproj, wprojT, 768, 768);
    // 3) QKV GEMM: [100352 x 768] @ [768 x 2304] -> bf16 qkv (+bias)
    gemm_bf16_kernel<true><<<dim3(9, 784), 256, 0, stream>>>(
        xb, wqkvT, bqkv, (void*)qkvb, 100352, 2304, 768);
    // 4) Windowed attention with fused q max-pool: 32*12*49 = 18816 waves
    attn_kernel<<<18816, 32, 0, stream>>>(qkvb, obuf);
    // 5) Proj GEMM: [25088 x 768] @ [768 x 768] -> fp32 out (+bias)
    gemm_bf16_kernel<false><<<dim3(3, 196), 256, 0, stream>>>(
        obuf, wprojT, bproj, (void*)out, 25088, 768, 768);
}
